// Attention_30185030156518
// MI455X (gfx1250) — compile-verified, matrix operand order checked by reading, not run on hardware
//
#include <hip/hip_runtime.h>
#include <hip/hip_bf16.h>

typedef __attribute__((ext_vector_type(16))) __bf16 v16bf;
typedef __attribute__((ext_vector_type(8)))  __bf16 v8bf;
typedef __attribute__((ext_vector_type(8)))  float  v8f;

static constexpr int B = 32, T = 2048, D = 512, U = 512;
static constexpr int TM = 128;        // rows (b,t) per block: 8 waves x 16
static constexpr int BSTRIDE = 520;   // LDS row stride (elems): 1040B -> 4-bank skew, conflict-free
static constexpr int LDSB_ELEMS = 16 * BSTRIDE;          // one staged W1 chunk

union BF16x16 { v16bf v; __bf16 e[16]; v8bf h[2]; };
union F8x32   { v8f v; float e[8]; };

#if __has_builtin(__builtin_amdgcn_tensor_load_to_lds) && __has_builtin(__builtin_amdgcn_s_wait_tensorcnt)
#define USE_TDM 1
#else
#define USE_TDM 0
#endif

// therock-10.0 HIP headers (amdgpu-toolchain, clang-23) ship the TDM header and
// use the 6-arg builtin; ROCm 7.2 (clang-22) has the 5-arg form.
#if __has_include(<hip/amd_detail/amd_gfx1250_TDM.h>)
#define TDM_SIX_ARGS 1
#else
#define TDM_SIX_ARGS 0
#endif

// Opaque 32-bit value + memory clobber. Used for the LDS buffer offset: the
// TDM DMA writes LDS behind the compiler's back, so the clobber keeps the
// B-fragment loads alive (round 3 folded them to undef) while leaving the
// pointer chain rooted at the __shared__ array so the loads stay ds_load_b128
// (round 4's pointer-escape degraded them to flat_load_b128 + >256 VGPRs).
__device__ __forceinline__ int opaque_i32(int x) {
  asm volatile("" : "+v"(x) : : "memory");
  return x;
}

// Branch-free tanh: e = exp(-2|x|) in (0,1] -> no overflow path needed.
__device__ __forceinline__ float fast_tanh(float x) {
  const float e = __expf(-2.0f * __builtin_fabsf(x));
  const float t = (1.0f - e) * __builtin_amdgcn_rcpf(1.0f + e);
  return __builtin_copysignf(t, x);
}

#if USE_TDM
typedef unsigned int v4u   __attribute__((ext_vector_type(4)));
typedef int          v8i32 __attribute__((ext_vector_type(8)));
typedef int          v4i32 __attribute__((ext_vector_type(4)));

// Issue one TDM DMA: 16 rows x 512 bf16 (16KB contiguous) -> LDS with 16B pad
// per 1024B row (row stride becomes BSTRIDE=520 elems). D# packed per ISA 8.3/8.4.
// Generic LDS pointers carry the LDS byte offset in their low 32 bits
// (flat->LDS mapping truncates to addr[31:0]).
__device__ __forceinline__ void tdm_stage(const __bf16* ldsdst, const __bf16* src) {
  const unsigned long long ga = (unsigned long long)(uintptr_t)src;
  v4u g0;
  g0.x = 1u;                                              // count=1, no gather
  g0.y = (unsigned)(uintptr_t)ldsdst;                     // lds_addr [63:32]
  g0.z = (unsigned)ga;                                    // global_addr [95:64]
  g0.w = (unsigned)((ga >> 32) & 0x1FFFFFFull) | (2u << 30);  // addr[56:32] | type=2
  v8i32 g1;
  g1[0] = (1 << 16)      // data_size = 1 -> 2 bytes
        | (1 << 20)      // pad_enable
        | (7 << 22)      // pad_interval = 7 -> 256 DWORDs (one 1024B row)
        | (3 << 25);     // pad_amount = 3 -> 4 DWORDs (16B)
  g1[1] = (int)(512u << 16);   // tensor_dim0[15:0]=512 (atomic_barrier_addr=0)
  g1[2] = (int)(512u << 16);   // tensor_dim0 hi=0; tensor_dim1[15:0]=512
  g1[3] = (int)(512u << 16);   // tensor_dim1 hi=0; tile_dim0=512
  g1[4] = 16;                  // tile_dim1=16; tile_dim2=0
  g1[5] = 512;                 // tensor_dim0_stride[31:0]=512
  g1[6] = 0;
  g1[7] = 0;
  const v4i32 z4 = {0, 0, 0, 0};
#if TDM_SIX_ARGS
  const v8i32 z8 = {0, 0, 0, 0, 0, 0, 0, 0};
  __builtin_amdgcn_tensor_load_to_lds(g0, g1, z4, z4, z8, 0);
#else
  __builtin_amdgcn_tensor_load_to_lds(g0, g1, z4, z4, 0);
#endif
}
#endif

// ---------------------------------------------------------------------------
// Kernel 0: W1t[u][d] = bf16(W1[d][u])  (one-time convert + transpose, 512KB)
// ---------------------------------------------------------------------------
__global__ __launch_bounds__(256) void w1_pack_kernel(
    const float* __restrict__ W1, __bf16* __restrict__ W1t) {
  const int idx = blockIdx.x * 256 + threadIdx.x;  // coalesced read d*U+u
  const int u = idx & (U - 1);
  const int d = idx >> 9;
  W1t[(size_t)u * D + d] = (__bf16)W1[idx];
}

// ---------------------------------------------------------------------------
// Kernel 1: bias[b,u] = b1[u] + b2[u] + sum_d last[b,d] * W2[d,u]
// ---------------------------------------------------------------------------
__global__ __launch_bounds__(256) void bias_kernel(
    const float* __restrict__ last, const float* __restrict__ W2,
    const float* __restrict__ b1, const float* __restrict__ b2,
    float* __restrict__ bias) {
  const int idx = blockIdx.x * 256 + threadIdx.x;      // b*U + u
  const int b = idx >> 9;
  const int u = idx & (U - 1);
  const float* lb = last + b * D;
  float acc = b1[u] + b2[u];
#pragma unroll 8
  for (int d = 0; d < D; ++d) acc += lb[d] * W2[(size_t)d * U + u];
  bias[idx] = acc;
}

// ---------------------------------------------------------------------------
// Kernel 2: scores[b,t] = sum_u tanh( full[b,t,:]@W1[:,u] + bias[b,u] ) * V[u]
// WMMA bf16 GEMM fused with bias+tanh+V reduction. One wave = 16 rows.
// W1 chunks staged to LDS by TDM DMA (double-buffered) or reg-prefetch fallback.
// ---------------------------------------------------------------------------
__global__ __launch_bounds__(256) void score_kernel(
    const float* __restrict__ full, const __bf16* __restrict__ W1t,
    const float* __restrict__ Vw, const float* __restrict__ bias,
    float* __restrict__ scores) {
#if USE_TDM
  __shared__ __bf16 ldsB[2][LDSB_ELEMS];               // double buffer, 33,280 B
#else
  __shared__ __bf16 ldsB[1][LDSB_ELEMS];
#endif

  const int tid  = threadIdx.x;
  const int wave = tid >> 5;
  const int lane = tid & 31;
  const int m    = lane & 15;    // A row within tile / B+C column
  const int hi   = lane >> 4;

  const int b  = blockIdx.x >> 4;                      // 16 blocks per batch
  const int t0 = (blockIdx.x & 15) * TM + wave * 16;   // wave's first row
  const int row = t0 + m;

  // ---- Preload A: 16 rows x 512 K, bf16, register resident (16 fragments) ----
  // 16-bit A 16x32 layout: lane m holds K = 8*hi + {0..7} and 16+8*hi + {0..7}
  const float* arow = full + (size_t)(b * T + row) * D;
  BF16x16 afrag[16];
#pragma unroll
  for (int kc = 0; kc < 16; ++kc) {
    const int k0 = kc * 32 + 8 * hi;
    const float4 x0 = ((const float4*)(arow + k0))[0];
    const float4 x1 = ((const float4*)(arow + k0))[1];
    const float4 y0 = ((const float4*)(arow + k0 + 16))[0];
    const float4 y1 = ((const float4*)(arow + k0 + 16))[1];
    afrag[kc].e[0]  = (__bf16)x0.x; afrag[kc].e[1]  = (__bf16)x0.y;
    afrag[kc].e[2]  = (__bf16)x0.z; afrag[kc].e[3]  = (__bf16)x0.w;
    afrag[kc].e[4]  = (__bf16)x1.x; afrag[kc].e[5]  = (__bf16)x1.y;
    afrag[kc].e[6]  = (__bf16)x1.z; afrag[kc].e[7]  = (__bf16)x1.w;
    afrag[kc].e[8]  = (__bf16)y0.x; afrag[kc].e[9]  = (__bf16)y0.y;
    afrag[kc].e[10] = (__bf16)y0.z; afrag[kc].e[11] = (__bf16)y0.w;
    afrag[kc].e[12] = (__bf16)y1.x; afrag[kc].e[13] = (__bf16)y1.y;
    afrag[kc].e[14] = (__bf16)y1.z; afrag[kc].e[15] = (__bf16)y1.w;
  }

  float sc[8] = {0.f, 0.f, 0.f, 0.f, 0.f, 0.f, 0.f, 0.f};

#if USE_TDM
  if (wave == 0) tdm_stage(&ldsB[0][0], W1t);          // prologue: stage chunk 0
#else
  // prologue: prefetch chunk 0 (16KB contiguous) into registers
  uint4 rb0, rb1, rb2, rb3;
  {
    const uint4* src = (const uint4*)W1t;
    rb0 = src[tid]; rb1 = src[tid + 256]; rb2 = src[tid + 512]; rb3 = src[tid + 768];
  }
#endif

  for (int nc = 0; nc < 32; ++nc) {                    // sweep U in 16-col chunks
    const int u0 = nc * 16;

#if USE_TDM
    __builtin_amdgcn_s_wait_tensorcnt(0);              // wave0: DMA for buf[nc&1] done
    __syncthreads();                                   // publish; prev readers done
    if (wave == 0 && nc + 1 < 32)                      // overlap next DMA with compute
      tdm_stage(&ldsB[(nc + 1) & 1][0], W1t + (size_t)(nc + 1) * (16 * D));
    // Opaque offset: keeps ds_loads alive + AS(3), ordered after the wait.
    const __bf16* bbuf = &ldsB[0][0] + opaque_i32((nc & 1) * LDSB_ELEMS);
#else
    __syncthreads();                                   // prev chunk's readers done
    {                                                  // regs -> padded LDS layout
      const int s0 = tid, s1 = tid + 256, s2 = tid + 512, s3 = tid + 768;
      *(uint4*)&ldsB[0][(s0 >> 6) * BSTRIDE + (s0 & 63) * 8] = rb0;
      *(uint4*)&ldsB[0][(s1 >> 6) * BSTRIDE + (s1 & 63) * 8] = rb1;
      *(uint4*)&ldsB[0][(s2 >> 6) * BSTRIDE + (s2 & 63) * 8] = rb2;
      *(uint4*)&ldsB[0][(s3 >> 6) * BSTRIDE + (s3 & 63) * 8] = rb3;
    }
    if (nc + 1 < 32) {                                 // prefetch next chunk
      const uint4* src = (const uint4*)(W1t + (size_t)(nc + 1) * (16 * D));
      rb0 = src[tid]; rb1 = src[tid + 256]; rb2 = src[tid + 512]; rb3 = src[tid + 768];
    }
    __syncthreads();                                   // staged chunk visible
    const __bf16* bbuf = &ldsB[0][0];
#endif

    v8f acc = {};
#pragma unroll
    for (int kc = 0; kc < 16; ++kc) {
      // B 32x16 fragment: column n = lane&15, K = 8*hi + {0..7}, 16+8*hi + {0..7}
      const __bf16* bp = bbuf + m * BSTRIDE + kc * 32 + 8 * hi;
      BF16x16 bf;
      bf.h[0] = *(const v8bf*)bp;          // ds_load_b128
      bf.h[1] = *(const v8bf*)(bp + 16);   // ds_load_b128
      acc = __builtin_amdgcn_wmma_f32_16x16x32_bf16(
          false, afrag[kc].v, false, bf.v, (short)0, acc, false, false);
    }

    // C layout: c[r] = C[row r + 8*hi][col = lane&15]
    const float bv = bias[b * U + u0 + m];
    const float vv = Vw[u0 + m];
    F8x32 c; c.v = acc;
#pragma unroll
    for (int r = 0; r < 8; ++r) sc[r] += fast_tanh(c.e[r] + bv) * vv;
  }

  // Reduce over the 16 columns held by lanes within each half (rows shared per half)
#pragma unroll
  for (int mask = 1; mask <= 8; mask <<= 1) {
#pragma unroll
    for (int r = 0; r < 8; ++r) sc[r] += __shfl_xor(sc[r], mask, 32);
  }

  if (m == 0) {                                        // lanes 0 and 16
#pragma unroll
    for (int r = 0; r < 8; ++r)
      scores[b * T + t0 + 8 * hi + r] = sc[r];
  }
}

// ---------------------------------------------------------------------------
// Kernel 3: per-batch softmax over T + context[b,d] = sum_t attn[t]*full[b,t,d]
// ---------------------------------------------------------------------------
__global__ __launch_bounds__(256) void softmax_ctx_kernel(
    const float* __restrict__ full, const float* __restrict__ scores,
    float* __restrict__ out) {
  __shared__ float w[T];
  __shared__ float red[256];
  const int b = blockIdx.x;
  const int tid = threadIdx.x;
  const float* sb = scores + b * T;

  float lmax = -3.402823466e38f;
  for (int t = tid; t < T; t += 256) { const float s = sb[t]; w[t] = s; lmax = fmaxf(lmax, s); }
  red[tid] = lmax; __syncthreads();
  for (int s = 128; s > 0; s >>= 1) {
    if (tid < s) red[tid] = fmaxf(red[tid], red[tid + s]);
    __syncthreads();
  }
  const float mx = red[0]; __syncthreads();

  float lsum = 0.f;
  for (int t = tid; t < T; t += 256) { const float e = __expf(w[t] - mx); w[t] = e; lsum += e; }
  red[tid] = lsum; __syncthreads();
  for (int s = 128; s > 0; s >>= 1) {
    if (tid < s) red[tid] += red[tid + s];
    __syncthreads();
  }
  const float inv = 1.0f / red[0];
  __syncthreads();

  // context: thread owns 2 consecutive d columns; 4-way unrolled over t.
  const float2* fb = (const float2*)(full + (size_t)b * T * D);
  float2 a0 = make_float2(0.f, 0.f), a1 = make_float2(0.f, 0.f);
  float2 a2 = make_float2(0.f, 0.f), a3 = make_float2(0.f, 0.f);
  for (int t = 0; t < T; t += 4) {
    const float w0 = w[t], w1 = w[t + 1], w2 = w[t + 2], w3 = w[t + 3];
    const float2 f0 = fb[(size_t)(t + 0) * 256 + tid];
    const float2 f1 = fb[(size_t)(t + 1) * 256 + tid];
    const float2 f2 = fb[(size_t)(t + 2) * 256 + tid];
    const float2 f3 = fb[(size_t)(t + 3) * 256 + tid];
    a0.x += w0 * f0.x; a0.y += w0 * f0.y;
    a1.x += w1 * f1.x; a1.y += w1 * f1.y;
    a2.x += w2 * f2.x; a2.y += w2 * f2.y;
    a3.x += w3 * f3.x; a3.y += w3 * f3.y;
  }
  const float rx = (a0.x + a1.x) + (a2.x + a3.x);
  const float ry = (a0.y + a1.y) + (a2.y + a3.y);
  out[b * D + tid * 2 + 0] = rx * inv;
  out[b * D + tid * 2 + 1] = ry * inv;
}

// ---------------------------------------------------------------------------
extern "C" void kernel_launch(void* const* d_in, const int* in_sizes, int n_in,
                              void* d_out, int out_size, void* d_ws, size_t ws_size,
                              hipStream_t stream) {
  const float* full = (const float*)d_in[0];   // [B,T,D]
  const float* last = (const float*)d_in[1];   // [B,D]
  const float* W1   = (const float*)d_in[2];   // [D,U]
  const float* b1   = (const float*)d_in[3];   // [U]
  const float* W2   = (const float*)d_in[4];   // [D,U]
  const float* b2   = (const float*)d_in[5];   // [U]
  const float* Vw   = (const float*)d_in[6];   // [U,1]
  // d_in[7] = bV: additive constant before softmax -> no effect on output.
  float* out = (float*)d_out;                  // [B,D]

  float*  bias   = (float*)d_ws;               // [B,U]    64 KB
  float*  scores = bias + B * U;               // [B,T]   256 KB
  __bf16* W1t    = (__bf16*)(scores + B * T);  // [U,D]   512 KB (bf16, transposed)

  w1_pack_kernel<<<(D * U) / 256, 256, 0, stream>>>(W1, W1t);
  bias_kernel<<<(B * U) / 256, 256, 0, stream>>>(last, W2, b1, b2, bias);
  score_kernel<<<(B * T) / TM, 256, 0, stream>>>(full, W1t, Vw, bias, scores);
  softmax_ctx_kernel<<<B, 256, 0, stream>>>(full, scores, out);
}
